// AtomAttentionEncoder_21852793602162
// MI455X (gfx1250) — matrix operand order repaired and checked by reading, not compiled, not run on hardware
//
#include <hip/hip_runtime.h>
#include <math.h>

typedef __attribute__((ext_vector_type(16))) _Float16     v16h;
typedef __attribute__((ext_vector_type(8)))  float        v8f;
typedef __attribute__((ext_vector_type(4)))  unsigned int u32x4;
typedef __attribute__((ext_vector_type(8)))  unsigned int u32x8;

#define NATOM 2048
#define CDIM  128
#define TCDIM 256
#define PADH  8   // LDS row padding in halfs (16B = 4 DWORDs, matches TDM pad)
#define LROW  (CDIM + PADH)   // 136 halfs per padded LDS row

// ---------------------------------------------------------------------------
// Tensor Data Mover: 2D tile (f16 elements) global -> LDS with row padding.
// tile_k = 128 halfs (256B = 64 DWORDs), pad 4 DWORDs -> 136-half LDS rows.
// Descriptor per CDNA5 ISA 8.3/8.4. Issued by one wave; EXEC ignored by TDM.
// ---------------------------------------------------------------------------
__device__ __forceinline__ void tdm_load_2d(unsigned lds_byte_off,
                                            unsigned long long gaddr,
                                            unsigned tile_rows,
                                            unsigned row_stride_elems) {
    u32x4 g0;
    g0[0] = 1u;                                            // count=1, user mode
    g0[1] = lds_byte_off;                                  // LDS byte address
    g0[2] = (unsigned)(gaddr & 0xFFFFFFFFull);             // global addr lo
    g0[3] = (unsigned)((gaddr >> 32) & 0x01FFFFFFull)      // global addr hi
          | (2u << 30);                                    // type = 2 (image)
    u32x8 g1;
    // wg_mask=0 | data_size=1(2B)<<16 | pad_enable<<20 | pad_interval=5(64dw)<<22
    // | pad_amount=3(4dw)<<25
    g1[0] = (1u << 16) | (1u << 20) | (5u << 22) | (3u << 25);
    g1[1] = (row_stride_elems & 0xFFFFu) << 16;            // tensor_dim0 lo
    g1[2] = ((row_stride_elems >> 16) & 0xFFFFu)           // tensor_dim0 hi
          | ((tile_rows & 0xFFFFu) << 16);                 // tensor_dim1 lo
    g1[3] = (128u << 16);                                  // tile_dim0 = 128 elems
    g1[4] = tile_rows & 0xFFFFu;                           // tile_dim1 (tile_dim2=0)
    g1[5] = row_stride_elems;                              // tensor_dim0_stride lo
    g1[6] = 0u;                                            // stride hi / dim1_stride
    g1[7] = 0u;
    asm volatile("tensor_load_to_lds %0, %1" :: "s"(g0), "s"(g1) : "memory");
}

// ---------------------------------------------------------------------------
// LayerNorm pair: an_pre = LN(a) (f32), sn = LN(s)*gamma (f16 for WMMA GEMMs)
// grid = NATOM, block = 128 (one row per block)
// ---------------------------------------------------------------------------
__global__ void ln_pair_kernel(const float* __restrict__ a, const float* __restrict__ s,
                               const float* __restrict__ gamma,
                               float* __restrict__ an_pre, _Float16* __restrict__ sn) {
    __shared__ float red[4][128];
    const int row = blockIdx.x;
    const int t   = threadIdx.x;
    float xa = a[(size_t)row * CDIM + t];
    float xs = s[(size_t)row * CDIM + t];
    red[0][t] = xa; red[1][t] = xa * xa;
    red[2][t] = xs; red[3][t] = xs * xs;
    __syncthreads();
    for (int off = 64; off > 0; off >>= 1) {
        if (t < off) {
            red[0][t] += red[0][t + off];
            red[1][t] += red[1][t + off];
            red[2][t] += red[2][t + off];
            red[3][t] += red[3][t + off];
        }
        __syncthreads();
    }
    const float inv = 1.0f / (float)CDIM;
    float ma = red[0][0] * inv, va = red[1][0] * inv - ma * ma;
    float ms = red[2][0] * inv, vs = red[3][0] * inv - ms * ms;
    an_pre[(size_t)row * CDIM + t] = (xa - ma) * rsqrtf(va + 1e-5f);
    sn[(size_t)row * CDIM + t] = (_Float16)((xs - ms) * rsqrtf(vs + 1e-5f) * gamma[t]);
}

// ---------------------------------------------------------------------------
// Weight convert: W[K][Nout] f32 -> Wt[Nout][K] f16 (transposed, TDM friendly)
// ---------------------------------------------------------------------------
__global__ void wt_convert(const float* __restrict__ W, _Float16* __restrict__ out,
                           int K, int Nout) {
    int i = blockIdx.x * 256 + threadIdx.x;
    if (i < K * Nout) {
        int k = i / Nout, n = i % Nout;
        out[(size_t)n * K + k] = (_Float16)W[i];
    }
}
__global__ void cvt_f2h(const float* __restrict__ x, _Float16* __restrict__ y, int n) {
    int i = blockIdx.x * 256 + threadIdx.x;
    if (i < n) y[i] = (_Float16)x[i];
}

// ---------------------------------------------------------------------------
// f16 WMMA GEMM with TDM staging: out = A[M,K] @ Wt[Nout,K]^T (+bias)
// A f16 row-major, Wt f16 transposed. grid=(M/128, Nout/128), block=256.
// ---------------------------------------------------------------------------
__global__ void gemm_f16_wmma(const _Float16* __restrict__ A, const _Float16* __restrict__ Wt,
                              const float* __restrict__ bias, float* __restrict__ outF,
                              _Float16* __restrict__ outH, int K, int Nout) {
    extern __shared__ _Float16 sm[];
    _Float16* sA = sm;                 // [128][LROW]  m x k
    _Float16* sW = sm + 128 * LROW;    // [128][LROW]  n x k

    const int tid  = threadIdx.x;
    const int lane = tid & 31;
    const int wv   = tid >> 5;
    const int l15  = lane & 15;
    const int lh   = lane >> 4;
    const int rowBase = blockIdx.x * 128;
    const int colBase = blockIdx.y * 128;

    v8f zero = {0.f, 0.f, 0.f, 0.f, 0.f, 0.f, 0.f, 0.f};
    v8f acc[8];
#pragma unroll
    for (int nt = 0; nt < 8; ++nt) acc[nt] = zero;

    for (int kt0 = 0; kt0 < K; kt0 += 128) {
        if (tid < 32) {   // wave 0 drives the DMA
            tdm_load_2d((unsigned)(unsigned long long)sA,
                        (unsigned long long)(A + (size_t)rowBase * K + kt0),
                        128u, (unsigned)K);
            tdm_load_2d((unsigned)(unsigned long long)sW,
                        (unsigned long long)(Wt + (size_t)colBase * K + kt0),
                        128u, (unsigned)K);
            __builtin_amdgcn_s_wait_tensorcnt(0);
        }
        __syncthreads();
#pragma unroll
        for (int kk = 0; kk < 4; ++kk) {
            v16h afrag;
#pragma unroll
            for (int e = 0; e < 16; ++e) {
                int k = kk * 32 + (e & 7) + ((e & 8) << 1) + (lh << 3);
                afrag[e] = sA[(wv * 16 + l15) * LROW + k];
            }
            v16h bfr[8];
#pragma unroll
            for (int nt = 0; nt < 8; ++nt)
#pragma unroll
                for (int e = 0; e < 16; ++e) {
                    int k = kk * 32 + (e & 7) + ((e & 8) << 1) + (lh << 3);
                    bfr[nt][e] = sW[(nt * 16 + l15) * LROW + k];
                }
#pragma unroll
            for (int nt = 0; nt < 8; ++nt)
                acc[nt] = __builtin_amdgcn_wmma_f32_16x16x32_f16(
                    false, afrag, false, bfr[nt], (short)0, acc[nt], false, false);
        }
        __syncthreads();
    }
#pragma unroll
    for (int nt = 0; nt < 8; ++nt) {
        int col = colBase + nt * 16 + l15;
        float bv = bias ? bias[col] : 0.f;
#pragma unroll
        for (int r = 0; r < 8; ++r) {
            int row = rowBase + wv * 16 + r + (lh << 3);
            float v = acc[nt][r] + bv;
            if (outF) outF[(size_t)row * Nout + col] = v;
            if (outH) outH[(size_t)row * Nout + col] = (_Float16)v;
        }
    }
}

// ---------------------------------------------------------------------------
// Fused local attention: one workgroup per 32-query window (64 windows).
// TDM stages Q/K/V (f16) while all waves compute the pair bias from plm.
// ---------------------------------------------------------------------------
__global__ void attn_kernel(const _Float16* __restrict__ q, const _Float16* __restrict__ kmat,
                            const _Float16* __restrict__ vmat, const _Float16* __restrict__ graw,
                            const float* __restrict__ plm, const float* __restrict__ lnz_g,
                            const float* __restrict__ lnz_b, const float* __restrict__ w_pair,
                            const float* __restrict__ amask, _Float16* __restrict__ o) {
    extern __shared__ unsigned char smraw[];
    _Float16* sQ  = (_Float16*)smraw;                 // [32][LROW]
    _Float16* sK  = sQ + 32 * LROW;                   // [128][LROW]
    _Float16* sV  = sK + 128 * LROW;                  // [128][LROW]
    float*    sPB = (float*)(sV + 128 * LROW);        // [4][32][128]
    _Float16* sP  = (_Float16*)(sPB + 4 * 32 * 128);  // [8][16][LROW]

    const int qb    = blockIdx.x;
    const int qbase = qb * 32;
    const int k0    = qbase - 48;
    const int tid  = threadIdx.x;
    const int lane = tid & 31;
    const int wv   = tid >> 5;
    const int l15  = lane & 15;
    const int lh   = lane >> 4;

    const int lo  = (k0 < 0) ? 0 : k0;
    const int hi  = (k0 + 128 > NATOM) ? NATOM : (k0 + 128);
    const int jlo = lo - k0, jhi = hi - k0;

    if (tid < 32) {   // wave 0: async DMA of Q window + clamped K/V windows
        tdm_load_2d((unsigned)(unsigned long long)sQ,
                    (unsigned long long)(q + (size_t)qbase * CDIM), 32u, CDIM);
        tdm_load_2d((unsigned)(unsigned long long)(sK + (size_t)jlo * LROW),
                    (unsigned long long)(kmat + (size_t)lo * CDIM),
                    (unsigned)(jhi - jlo), CDIM);
        tdm_load_2d((unsigned)(unsigned long long)(sV + (size_t)jlo * LROW),
                    (unsigned long long)(vmat + (size_t)lo * CDIM),
                    (unsigned)(jhi - jlo), CDIM);
    }
    // zero-fill out-of-range K/V rows (disjoint from TDM target rows)
    for (int idx = tid; idx < 128 * CDIM; idx += 256) {
        int j = idx >> 7, c = idx & 127;
        if (j < jlo || j >= jhi) {
            sK[j * LROW + c] = (_Float16)0.f;
            sV[j * LROW + c] = (_Float16)0.f;
        }
    }
    // pair bias: LN(plm)*g+b @ w_pair, + key validity / mask bias (overlaps TDM)
    for (int p = tid; p < 32 * 128; p += 256) {
        int qr = p >> 7, j = p & 127;
        int kidx = k0 + j;
        float h0 = 0.f, h1 = 0.f, h2 = 0.f, h3 = 0.f, bv = -1e9f;
        if (kidx >= 0 && kidx < NATOM) {
            const float* z = plm + ((size_t)(qbase + qr) * NATOM + (size_t)kidx) * 16;
            float zz[16]; float m = 0.f;
#pragma unroll
            for (int c = 0; c < 16; ++c) { zz[c] = z[c]; m += zz[c]; }
            m *= (1.f / 16.f);
            float var = 0.f;
#pragma unroll
            for (int c = 0; c < 16; ++c) { float d = zz[c] - m; var += d * d; }
            var *= (1.f / 16.f);
            float inv = rsqrtf(var + 1e-5f);
#pragma unroll
            for (int c = 0; c < 16; ++c) {
                float y = (zz[c] - m) * inv * lnz_g[c] + lnz_b[c];
                h0 += y * w_pair[c * 4 + 0];
                h1 += y * w_pair[c * 4 + 1];
                h2 += y * w_pair[c * 4 + 2];
                h3 += y * w_pair[c * 4 + 3];
            }
            bv = (amask[kidx] - 1.f) * 1e9f;
        }
        sPB[(0 * 32 + qr) * 128 + j] = h0 + bv;
        sPB[(1 * 32 + qr) * 128 + j] = h1 + bv;
        sPB[(2 * 32 + qr) * 128 + j] = h2 + bv;
        sPB[(3 * 32 + qr) * 128 + j] = h3 + bv;
    }
    if (tid < 32) __builtin_amdgcn_s_wait_tensorcnt(0);
    __syncthreads();

    const int h  = wv >> 1;
    const int m0 = (wv & 1) * 16;
    const int hc = h * 32;

    // logits = Qh @ Kh^T : one WMMA (K=D=32) per 16-key tile
    v8f zero = {0.f, 0.f, 0.f, 0.f, 0.f, 0.f, 0.f, 0.f};
    v8f lacc[8];
#pragma unroll
    for (int nt = 0; nt < 8; ++nt) lacc[nt] = zero;
    {
        v16h afrag;
#pragma unroll
        for (int e = 0; e < 16; ++e) {
            int d = (e & 7) + ((e & 8) << 1) + (lh << 3);
            afrag[e] = sQ[(m0 + l15) * LROW + hc + d];
        }
        v16h bfr[8];
#pragma unroll
        for (int nt = 0; nt < 8; ++nt)
#pragma unroll
            for (int e = 0; e < 16; ++e) {
                int d = (e & 7) + ((e & 8) << 1) + (lh << 3);
                bfr[nt][e] = sK[(nt * 16 + l15) * LROW + hc + d];
            }
#pragma unroll
        for (int nt = 0; nt < 8; ++nt)
            lacc[nt] = __builtin_amdgcn_wmma_f32_16x16x32_f16(
                false, afrag, false, bfr[nt], (short)0, lacc[nt], false, false);
    }

    // softmax over 128 keys per q-row (rows live on 16-lane halves; wave32 shfl)
    const float scale = 0.1767766952966369f; // 1/sqrt(32)
    float lv[8][8], mx[8], sum[8];
#pragma unroll
    for (int r = 0; r < 8; ++r) mx[r] = -3.0e38f;
#pragma unroll
    for (int nt = 0; nt < 8; ++nt)
#pragma unroll
        for (int r = 0; r < 8; ++r) {
            int qr = m0 + r + lh * 8;
            float l = lacc[nt][r] * scale + sPB[(h * 32 + qr) * 128 + nt * 16 + l15];
            lv[nt][r] = l;
            mx[r] = fmaxf(mx[r], l);
        }
#pragma unroll
    for (int r = 0; r < 8; ++r)
        for (int off = 1; off <= 8; off <<= 1)
            mx[r] = fmaxf(mx[r], __shfl_xor(mx[r], off, 32));
#pragma unroll
    for (int r = 0; r < 8; ++r) sum[r] = 0.f;
#pragma unroll
    for (int nt = 0; nt < 8; ++nt)
#pragma unroll
        for (int r = 0; r < 8; ++r) {
            float pv = expf(lv[nt][r] - mx[r]);
            lv[nt][r] = pv;
            sum[r] += pv;
        }
#pragma unroll
    for (int r = 0; r < 8; ++r)
        for (int off = 1; off <= 8; off <<= 1)
            sum[r] += __shfl_xor(sum[r], off, 32);
#pragma unroll
    for (int nt = 0; nt < 8; ++nt)
#pragma unroll
        for (int r = 0; r < 8; ++r)
            sP[(wv * 16 + r + lh * 8) * LROW + nt * 16 + l15] =
                (_Float16)(lv[nt][r] / sum[r]);
    __syncthreads();

    // O = P @ Vh : 4 key-steps of 32, 2 column tiles (D=32)
    v8f oacc[2];
    oacc[0] = zero; oacc[1] = zero;
#pragma unroll
    for (int kk = 0; kk < 4; ++kk) {
        v16h pfrag;
#pragma unroll
        for (int e = 0; e < 16; ++e) {
            int key = kk * 32 + (e & 7) + ((e & 8) << 1) + (lh << 3);
            pfrag[e] = sP[(wv * 16 + l15) * LROW + key];
        }
        v16h vfr[2];
#pragma unroll
        for (int dt = 0; dt < 2; ++dt)
#pragma unroll
            for (int e = 0; e < 16; ++e) {
                int key = kk * 32 + (e & 7) + ((e & 8) << 1) + (lh << 3);
                vfr[dt][e] = sV[key * LROW + hc + dt * 16 + l15];
            }
#pragma unroll
        for (int dt = 0; dt < 2; ++dt)
            oacc[dt] = __builtin_amdgcn_wmma_f32_16x16x32_f16(
                false, pfrag, false, vfr[dt], (short)0, oacc[dt], false, false);
    }
    // gated epilogue: o = sigmoid(an@w_gate) * attn_out   (f16 out for next GEMM)
#pragma unroll
    for (int dt = 0; dt < 2; ++dt)
#pragma unroll
        for (int r = 0; r < 8; ++r) {
            int row = qbase + m0 + r + lh * 8;
            int col = hc + dt * 16 + l15;
            float gg = (float)graw[(size_t)row * CDIM + col];
            float sg = 1.f / (1.f + expf(-gg));
            o[(size_t)row * CDIM + col] = (_Float16)(sg * oacc[dt][r]);
        }
}

// ---------------------------------------------------------------------------
// Elementwise kernels
// ---------------------------------------------------------------------------
__global__ void ew_adaln(const float* __restrict__ x1, const float* __restrict__ x2,
                         const float* __restrict__ an_pre, _Float16* __restrict__ an, int n) {
    int i = blockIdx.x * blockDim.x + threadIdx.x;
    if (i < n) {
        float sg = 1.f / (1.f + expf(-x1[i]));
        an[i] = (_Float16)(sg * an_pre[i] + x2[i]);
    }
}
__global__ void ew_sig_mul(const float* __restrict__ g, const float* __restrict__ x,
                           float* __restrict__ out, int n) {
    int i = blockIdx.x * blockDim.x + threadIdx.x;
    if (i < n) out[i] = (1.f / (1.f + expf(-g[i]))) * x[i];
}
__global__ void ew_silu_mul(const float* __restrict__ t1, const float* __restrict__ t2,
                            _Float16* __restrict__ t, int n) {
    int i = blockIdx.x * blockDim.x + threadIdx.x;
    if (i < n) {
        float x = t1[i];
        t[i] = (_Float16)((x / (1.f + expf(-x))) * t2[i]);
    }
}
__global__ void ew_final(const float* __restrict__ battn, const float* __restrict__ g,
                         const float* __restrict__ t3, float* __restrict__ aOut,
                         float* __restrict__ out2, int n) {
    int i = blockIdx.x * blockDim.x + threadIdx.x;
    if (i < n) {
        float v = battn[i] + (1.f / (1.f + expf(-g[i]))) * t3[i];
        aOut[i] = v;
        if (out2) out2[i] = v;
    }
}

// ---------------------------------------------------------------------------
// Host orchestration
// ---------------------------------------------------------------------------
extern "C" void kernel_launch(void* const* d_in, const int* in_sizes, int n_in,
                              void* d_out, int out_size, void* d_ws, size_t ws_size,
                              hipStream_t stream) {
    const int N = NATOM, C = CDIM, TC = TCDIM, HD = 128, CP = 16, H = 4;

    const float* ql        = (const float*)d_in[0];
    const float* cl        = (const float*)d_in[1];
    const float* plm       = (const float*)d_in[2];
    const float* amask     = (const float*)d_in[3];
    const float* attn_gamma= (const float*)d_in[4];
    const float* attn_wsig = (const float*)d_in[5];
    const float* attn_bsig = (const float*)d_in[6];
    const float* attn_wskip= (const float*)d_in[7];
    const float* wq        = (const float*)d_in[8];
    const float* bq        = (const float*)d_in[9];
    const float* wk        = (const float*)d_in[10];
    const float* wv        = (const float*)d_in[11];
    const float* lnz_g     = (const float*)d_in[12];
    const float* lnz_b     = (const float*)d_in[13];
    const float* w_pair    = (const float*)d_in[14];
    const float* w_gate    = (const float*)d_in[15];
    const float* w_out     = (const float*)d_in[16];
    const float* w_ao      = (const float*)d_in[17];
    const float* b_ao      = (const float*)d_in[18];
    const float* tr_gamma  = (const float*)d_in[19];
    const float* tr_wsig   = (const float*)d_in[20];
    const float* tr_bsig   = (const float*)d_in[21];
    const float* tr_wskip  = (const float*)d_in[22];
    const float* w_t1      = (const float*)d_in[23];
    const float* w_t2      = (const float*)d_in[24];
    const float* w_t3      = (const float*)d_in[25];
    const float* w_to      = (const float*)d_in[26];
    const float* b_to      = (const float*)d_in[27];

    // ---- workspace carve ----
    char* p = (char*)d_ws;
    auto allocB = [&](size_t bytes) { char* r = p; p += (bytes + 255) & ~(size_t)255; return r; };
    float*    aBuf   = (float*)allocB((size_t)N * C * 4);
    float*    an_pre = (float*)allocB((size_t)N * C * 4);
    float*    tmp1   = (float*)allocB((size_t)N * TC * 4);
    float*    tmp2   = (float*)allocB((size_t)N * TC * 4);
    float*    battn  = (float*)allocB((size_t)N * C * 4);
    _Float16* snb    = (_Float16*)allocB((size_t)N * C * 2);
    _Float16* anb    = (_Float16*)allocB((size_t)N * C * 2);
    _Float16* qh     = (_Float16*)allocB((size_t)N * C * 2);
    _Float16* kh     = (_Float16*)allocB((size_t)N * C * 2);
    _Float16* vh     = (_Float16*)allocB((size_t)N * C * 2);
    _Float16* gh     = (_Float16*)allocB((size_t)N * C * 2);
    _Float16* oh     = (_Float16*)allocB((size_t)N * C * 2);
    _Float16* clh    = (_Float16*)allocB((size_t)N * C * 2);
    _Float16* th     = (_Float16*)allocB((size_t)N * TC * 2);
    // transposed f16 weights (3 blocks each)
    const size_t S128 = (size_t)C * 128, S256 = (size_t)C * 256;
    _Float16* w_sig_h  = (_Float16*)allocB(3 * S128 * 2);
    _Float16* w_skip_h = (_Float16*)allocB(3 * S128 * 2);
    _Float16* wq_h     = (_Float16*)allocB(3 * S128 * 2);
    _Float16* wk_h     = (_Float16*)allocB(3 * S128 * 2);
    _Float16* wv_h     = (_Float16*)allocB(3 * S128 * 2);
    _Float16* wg_h     = (_Float16*)allocB(3 * S128 * 2);
    _Float16* wo_h     = (_Float16*)allocB(3 * S128 * 2);
    _Float16* wao_h    = (_Float16*)allocB(3 * S128 * 2);
    _Float16* twsig_h  = (_Float16*)allocB(3 * S128 * 2);
    _Float16* twskip_h = (_Float16*)allocB(3 * S128 * 2);
    _Float16* wto_h    = (_Float16*)allocB(3 * S128 * 2);
    _Float16* wt1_h    = (_Float16*)allocB(3 * S256 * 2);
    _Float16* wt2_h    = (_Float16*)allocB(3 * S256 * 2);
    _Float16* wt3_h    = (_Float16*)allocB(3 * S256 * 2);

    const size_t gemmSmem = (size_t)2 * 128 * LROW * 2;                       // ~68KB
    const size_t attnSmem = (size_t)(32 * LROW + 2 * 128 * LROW + 8 * 16 * LROW) * 2
                          + (size_t)4 * 32 * 128 * 4;                          // ~175KB
    hipFuncSetAttribute((const void*)gemm_f16_wmma,
                        hipFuncAttributeMaxDynamicSharedMemorySize, (int)gemmSmem);
    hipFuncSetAttribute((const void*)attn_kernel,
                        hipFuncAttributeMaxDynamicSharedMemorySize, (int)attnSmem);

    // seed residual stream + f16 copies of static tensors
    hipMemcpyAsync(aBuf, ql, (size_t)N * C * sizeof(float), hipMemcpyDeviceToDevice, stream);
    cvt_f2h<<<(N * C + 255) / 256, 256, 0, stream>>>(cl, clh, N * C);
    auto conv = [&](const float* src, _Float16* dst, int K, int Nout) {
        size_t per = (size_t)K * Nout;
        for (int i = 0; i < 3; ++i)
            wt_convert<<<(int)((per + 255) / 256), 256, 0, stream>>>(
                src + i * per, dst + i * per, K, Nout);
    };
    conv(attn_wsig, w_sig_h, C, C);   conv(attn_wskip, w_skip_h, C, C);
    conv(wq, wq_h, C, HD);            conv(wk, wk_h, C, HD);
    conv(wv, wv_h, C, HD);            conv(w_gate, wg_h, C, HD);
    conv(w_out, wo_h, HD, C);         conv(w_ao, wao_h, C, C);
    conv(tr_wsig, twsig_h, C, C);     conv(tr_wskip, twskip_h, C, C);
    conv(w_to, wto_h, C, C);          conv(w_t1, wt1_h, C, TC);
    conv(w_t2, wt2_h, C, TC);         conv(w_t3, wt3_h, TC, C);

    const int ewC  = (N * C + 255) / 256;
    const int ewTC = (N * TC + 255) / 256;
    dim3 g128(16, 1), g256(16, 2);

    for (int i = 0; i < 3; ++i) {
        const _Float16* p_wsg  = w_sig_h  + (size_t)i * S128;
        const _Float16* p_wsk  = w_skip_h + (size_t)i * S128;
        const _Float16* p_wq   = wq_h + (size_t)i * S128;
        const _Float16* p_wk   = wk_h + (size_t)i * S128;
        const _Float16* p_wv   = wv_h + (size_t)i * S128;
        const _Float16* p_wg   = wg_h + (size_t)i * S128;
        const _Float16* p_wo   = wo_h + (size_t)i * S128;
        const _Float16* p_wao  = wao_h + (size_t)i * S128;
        const _Float16* p_twsg = twsig_h + (size_t)i * S128;
        const _Float16* p_twsk = twskip_h + (size_t)i * S128;
        const _Float16* p_wto  = wto_h + (size_t)i * S128;
        const _Float16* p_wt1  = wt1_h + (size_t)i * S256;
        const _Float16* p_wt2  = wt2_h + (size_t)i * S256;
        const _Float16* p_wt3  = wt3_h + (size_t)i * S256;

        // --- attention branch ---
        ln_pair_kernel<<<N, 128, 0, stream>>>(aBuf, cl, attn_gamma + (size_t)i * C, an_pre, snb);
        gemm_f16_wmma<<<g128, 256, gemmSmem, stream>>>(snb, p_wsg, attn_bsig + (size_t)i * C,
                                                       tmp1, nullptr, 128, 128);
        gemm_f16_wmma<<<g128, 256, gemmSmem, stream>>>(snb, p_wsk, nullptr, tmp2, nullptr, 128, 128);
        ew_adaln<<<ewC, 256, 0, stream>>>(tmp1, tmp2, an_pre, anb, N * C);
        gemm_f16_wmma<<<g128, 256, gemmSmem, stream>>>(anb, p_wq, bq + (size_t)i * HD,
                                                       nullptr, qh, 128, 128);
        gemm_f16_wmma<<<g128, 256, gemmSmem, stream>>>(anb, p_wk, nullptr, nullptr, kh, 128, 128);
        gemm_f16_wmma<<<g128, 256, gemmSmem, stream>>>(anb, p_wv, nullptr, nullptr, vh, 128, 128);
        gemm_f16_wmma<<<g128, 256, gemmSmem, stream>>>(anb, p_wg, nullptr, nullptr, gh, 128, 128);
        attn_kernel<<<64, 256, attnSmem, stream>>>(qh, kh, vh, gh, plm,
                                                   lnz_g + (size_t)i * CP, lnz_b + (size_t)i * CP,
                                                   w_pair + (size_t)i * CP * H, amask, oh);
        gemm_f16_wmma<<<g128, 256, gemmSmem, stream>>>(oh, p_wo, nullptr, tmp1, nullptr, 128, 128);
        gemm_f16_wmma<<<g128, 256, gemmSmem, stream>>>(clh, p_wao, b_ao + (size_t)i * C,
                                                       tmp2, nullptr, 128, 128);
        ew_sig_mul<<<ewC, 256, 0, stream>>>(tmp2, tmp1, battn, N * C);

        // --- transition branch (uses old a) ---
        ln_pair_kernel<<<N, 128, 0, stream>>>(aBuf, cl, tr_gamma + (size_t)i * C, an_pre, snb);
        gemm_f16_wmma<<<g128, 256, gemmSmem, stream>>>(snb, p_twsg, tr_bsig + (size_t)i * C,
                                                       tmp1, nullptr, 128, 128);
        gemm_f16_wmma<<<g128, 256, gemmSmem, stream>>>(snb, p_twsk, nullptr, tmp2, nullptr, 128, 128);
        ew_adaln<<<ewC, 256, 0, stream>>>(tmp1, tmp2, an_pre, anb, N * C);
        gemm_f16_wmma<<<g256, 256, gemmSmem, stream>>>(anb, p_wt1, nullptr, tmp1, nullptr, 128, 256);
        gemm_f16_wmma<<<g256, 256, gemmSmem, stream>>>(anb, p_wt2, nullptr, tmp2, nullptr, 128, 256);
        ew_silu_mul<<<ewTC, 256, 0, stream>>>(tmp1, tmp2, th, N * TC);
        gemm_f16_wmma<<<g128, 256, gemmSmem, stream>>>(th, p_wt3, nullptr, tmp1, nullptr, 256, 128);
        gemm_f16_wmma<<<g128, 256, gemmSmem, stream>>>(clh, p_wto, b_to + (size_t)i * C,
                                                       tmp2, nullptr, 128, 128);
        ew_final<<<ewC, 256, 0, stream>>>(battn, tmp2, tmp1, aBuf,
                                          (i == 2) ? (float*)d_out : nullptr, N * C);
    }
}
